// PGM_40827959116157
// MI455X (gfx1250) — matrix operand order repaired and checked by reading, not compiled, not run on hardware
//
#include <hip/hip_runtime.h>
#include <hip/hip_bf16.h>
#include <math.h>

#define NPTS   65536
#define DF     256
#define NANC   128
#define KN     1024
#define HID    512
#define NBINS  12
#define KP1    288   // 259 padded to multiple of 32 (layout: [feat 0..255 | xyz 256..258 | 0])
#define MCH    32    // M-chunk rows per anchor-MLP iteration

typedef __attribute__((ext_vector_type(16))) _Float16 v16h;
typedef __attribute__((ext_vector_type(8)))  _Float16 v8h;
typedef __attribute__((ext_vector_type(8)))  float    v8f;

// ---------------- WMMA fragment loaders (gfx1250 wave32 layouts) ----------------
// A (16x32 f16, row-major, lda in halves): lane 0-15 -> M=lane, K = {0..7, 16..23}
//                                          lane 16-31 -> M=lane-16, K = {8..15, 24..31}
__device__ __forceinline__ v16h load_a16(const _Float16* A, int lda) {
    int lane = threadIdx.x & 31;
    int hh = lane >> 4, m = lane & 15;
    const _Float16* p0 = A + m * lda + 8 * hh;
    const _Float16* p1 = A + m * lda + 16 + 8 * hh;
    v8h lo = *(const v8h*)p0;
    v8h hi = *(const v8h*)p1;
    v16h r;
#pragma unroll
    for (int i = 0; i < 8; ++i) { r[i] = lo[i]; r[i + 8] = hi[i]; }
    return r;
}
// B (32x16 f16) from a COLUMN-MAJOR weight matrix Bt (N x K, ldk in halves):
// lane 0-15 -> N=lane, K = {0..7, 16..23} ; lane 16-31 -> N=lane-16, K = {8..15, 24..31}
__device__ __forceinline__ v16h load_b16(const _Float16* BtK0, int ldk) {
    int lane = threadIdx.x & 31;
    int hh = lane >> 4, n = lane & 15;
    const _Float16* p = BtK0 + n * ldk + 8 * hh;
    v8h lo = *(const v8h*)p;
    v8h hi = *(const v8h*)(p + 16);
    v16h r;
#pragma unroll
    for (int i = 0; i < 8; ++i) { r[i] = lo[i]; r[i + 8] = hi[i]; }
    return r;
}

__device__ __forceinline__ v8f wmma_f16(v16h a, v16h b, v8f c) {
    return __builtin_amdgcn_wmma_f32_16x16x32_f16(false, a, false, b, (short)0, c,
                                                  false, false);
}

// ---------------- weight prep: f32 (KxN) -> f16 column-major (NxKpad) -----------
__global__ void k_prep_wt(const float* __restrict__ W, _Float16* __restrict__ Wt,
                          int K, int N, int Kpad) {
    int n = blockIdx.x;
    for (int k = threadIdx.x; k < Kpad; k += blockDim.x)
        Wt[n * Kpad + k] = (k < K) ? (_Float16)W[k * N + n] : (_Float16)0.f;
}
// Wp1 with K permuted to [feat rows 3..258 | xyz rows 0..2 | zero pad]
__global__ void k_prep_wp1(const float* __restrict__ W, _Float16* __restrict__ Wt) {
    int n = blockIdx.x;
    for (int k = threadIdx.x; k < KP1; k += blockDim.x) {
        float v;
        if (k < 256)      v = W[(3 + k) * HID + n];
        else if (k < 259) v = W[(k - 256) * HID + n];
        else              v = 0.f;
        Wt[n * KP1 + k] = (_Float16)v;
    }
}

// ---------------- h = relu(points @ W1 + b1), K=4 (VALU) ----------------
__global__ void k_h(const float* __restrict__ pts, const float* __restrict__ W1,
                    const float* __restrict__ b1, _Float16* __restrict__ h16) {
    int i = blockIdx.x, j = threadIdx.x;
    const float* p = pts + i * 4;
    float acc = b1[j];
#pragma unroll
    for (int k = 0; k < 4; ++k) acc += p[k] * W1[k * DF + j];
    h16[i * DF + j] = (_Float16)fmaxf(acc, 0.f);
}

// ---------------- feat = relu(h @ W2 + b2), 2x2 register-blocked WMMA -----------
__global__ void k_feat(const _Float16* __restrict__ h16, const _Float16* __restrict__ W2t,
                       const float* __restrict__ b2,
                       float* __restrict__ featOut, _Float16* __restrict__ feat16) {
    int wave = threadIdx.x >> 5;
    int tile = blockIdx.x * 8 + wave;        // 32x32 macro-tile per wave; 16384 total
    int mt2 = tile >> 3, nt2 = tile & 7;     // (65536/32) x (256/32)
    int lane = threadIdx.x & 31, hh = lane >> 4, nl = lane & 15;
    int n0 = nt2 * 32 + nl, n1 = n0 + 16;
    float bv0 = b2[n0], bv1 = b2[n1];
    v8f acc00, acc01, acc10, acc11;
#pragma unroll
    for (int r = 0; r < 8; ++r) { acc00[r] = bv0; acc01[r] = bv1; acc10[r] = bv0; acc11[r] = bv1; }
    const _Float16* Ab = h16 + (size_t)(mt2 * 32) * DF;
    const _Float16* Bb = W2t + (size_t)(nt2 * 32) * DF;
#pragma unroll
    for (int kt = 0; kt < DF / 32; ++kt) {
        v16h a0 = load_a16(Ab + kt * 32, DF);
        v16h a1 = load_a16(Ab + 16 * DF + kt * 32, DF);
        v16h b0 = load_b16(Bb + kt * 32, DF);
        v16h b1 = load_b16(Bb + 16 * DF + kt * 32, DF);
        acc00 = wmma_f16(a0, b0, acc00);
        acc01 = wmma_f16(a0, b1, acc01);
        acc10 = wmma_f16(a1, b0, acc10);
        acc11 = wmma_f16(a1, b1, acc11);
    }
#pragma unroll
    for (int r = 0; r < 8; ++r) {
        int m0 = mt2 * 32 + r + 8 * hh, m1 = m0 + 16;
        float v00 = fmaxf(acc00[r], 0.f), v01 = fmaxf(acc01[r], 0.f);
        float v10 = fmaxf(acc10[r], 0.f), v11 = fmaxf(acc11[r], 0.f);
        featOut[m0 * DF + n0] = v00; feat16[m0 * DF + n0] = (_Float16)v00;
        featOut[m0 * DF + n1] = v01; feat16[m0 * DF + n1] = (_Float16)v01;
        featOut[m1 * DF + n0] = v10; feat16[m1 * DF + n0] = (_Float16)v10;
        featOut[m1 * DF + n1] = v11; feat16[m1 * DF + n1] = (_Float16)v11;
    }
}

// ---------------- score = sigmoid(feat @ Ws + bs) ----------------
__global__ void k_score(const float* __restrict__ feat, const float* __restrict__ Ws,
                        const float* __restrict__ bs, float* __restrict__ score) {
    int i = blockIdx.x * blockDim.x + threadIdx.x;
    float acc = bs[0];
    const float* f = feat + i * DF;
    for (int k = 0; k < DF; ++k) acc += f[k] * Ws[k];
    score[i] = 1.f / (1.f + expf(-acc));
}

// ---------------- NMS: 128 rounds of argmax + radius suppression (1 block) ------
__global__ void k_nms(float* __restrict__ score, const float* __restrict__ pts,
                      int* __restrict__ anc_idx, float* __restrict__ anc_xyz) {
    __shared__ float bv[1024];
    __shared__ int   bi[1024];
    __shared__ float c[3];
    int t = threadIdx.x;
    for (int it = 0; it < NANC; ++it) {
        float mv = -INFINITY; int mi = 0x7fffffff;
        for (int i = t; i < NPTS; i += 1024) {
            float v = score[i];
            if (v > mv || (v == mv && i < mi)) { mv = v; mi = i; }
        }
        bv[t] = mv; bi[t] = mi;
        __syncthreads();
        for (int s = 512; s > 0; s >>= 1) {
            if (t < s) {
                float ov = bv[t + s]; int oi = bi[t + s];
                if (ov > bv[t] || (ov == bv[t] && oi < bi[t])) { bv[t] = ov; bi[t] = oi; }
            }
            __syncthreads();
        }
        if (t == 0) {
            int idx = bi[0];
            anc_idx[it] = idx;
            c[0] = pts[idx * 4]; c[1] = pts[idx * 4 + 1]; c[2] = pts[idx * 4 + 2];
            anc_xyz[it * 4 + 0] = c[0]; anc_xyz[it * 4 + 1] = c[1];
            anc_xyz[it * 4 + 2] = c[2]; anc_xyz[it * 4 + 3] = 0.f;
        }
        __syncthreads();
        float cx = c[0], cy = c[1], cz = c[2];
        for (int i = t; i < NPTS; i += 1024) {
            float dx = pts[i * 4] - cx, dy = pts[i * 4 + 1] - cy, dz = pts[i * 4 + 2] - cz;
            if (dx * dx + dy * dy + dz * dz < 4.0f) score[i] = -INFINITY;
        }
        __syncthreads();
    }
}

// ---------------- deterministic neighbor compaction (d2 <= 16, cap 1024) --------
__global__ void k_select(const float* __restrict__ pts, const float* __restrict__ anc_xyz,
                         int* __restrict__ nidx, int* __restrict__ ncnt) {
    int a = blockIdx.x;
    __shared__ int base;
    __shared__ int wsum[8];
    if (threadIdx.x == 0) base = 0;
    __syncthreads();
    float cx = anc_xyz[a * 4], cy = anc_xyz[a * 4 + 1], cz = anc_xyz[a * 4 + 2];
    int lane = threadIdx.x & 31, wv = threadIdx.x >> 5;
    unsigned long long lmask = (1ull << lane) - 1ull;
    for (int start = 0; start < NPTS; start += 256) {
        int i = start + threadIdx.x;
        float dx = pts[i * 4] - cx, dy = pts[i * 4 + 1] - cy, dz = pts[i * 4 + 2] - cz;
        bool f = (dx * dx + dy * dy + dz * dz) <= 16.0f;
        unsigned long long bal = __ballot(f);
        if (lane == 0) wsum[wv] = __popcll(bal);
        __syncthreads();
        int woff = 0;
        for (int j = 0; j < wv; ++j) woff += wsum[j];
        if (f) {
            int pos = base + woff + __popcll(bal & lmask);
            if (pos < KN) nidx[a * KN + pos] = i;
        }
        __syncthreads();
        if (threadIdx.x == 0) {
            int tot = 0;
            for (int j = 0; j < 8; ++j) tot += wsum[j];
            base += tot;
        }
        __syncthreads();
    }
    if (threadIdx.x == 0) ncnt[a] = base < KN ? base : KN;
}

// ---------------- per-anchor 2-layer MLP (WMMA, async gather) + max-pool --------
__global__ void k_mlp(const float* __restrict__ pts, const _Float16* __restrict__ feat16,
                      const float* __restrict__ anc_xyz,
                      const int* __restrict__ nidx, const int* __restrict__ ncnt,
                      const _Float16* __restrict__ Wp1t, const float* __restrict__ bp1,
                      const _Float16* __restrict__ Wp2t, const float* __restrict__ bp2,
                      float* __restrict__ pooledOut) {
    extern __shared__ char smem[];
    _Float16* A1 = (_Float16*)smem;                               // MCH x KP1  (offset 0)
    _Float16* G1 = (_Float16*)(smem + MCH * KP1 * 2);             // MCH x HID
    float* pool = (float*)(smem + MCH * KP1 * 2 + MCH * HID * 2); // HID
    int a = blockIdx.x, t = threadIdx.x;
    int cnt = ncnt[a];
    float cx = anc_xyz[a * 4], cy = anc_xyz[a * 4 + 1], cz = anc_xyz[a * 4 + 2];
    pool[t] = 0.f; pool[t + 256] = 0.f;
    int wv = t >> 5, lane = t & 31, hh = lane >> 4, nl = lane & 15;
    int nch = (cnt + MCH - 1) / MCH;
    for (int ch = 0; ch < nch; ++ch) {
        __syncthreads();
        { // gather A tile: [feat row (async DMA) | local xyz | zero pad]
            int r = t >> 3, g = t & 7;          // 32 rows x 8 threads/row
            int row = ch * MCH + r;
            bool valid = row < cnt;
            int idx = valid ? nidx[a * KN + row] : 0;
            // tail columns 256..287 (xyz + pad): 4 halves per thread
            const float* p = pts + idx * 4;
            float lxyz[3] = { p[0] - cx, p[1] - cy, p[2] - cz };
#pragma unroll
            for (int j = 0; j < 4; ++j) {
                int c = 256 + g * 4 + j;
                _Float16 v = (_Float16)0.f;
                if (valid && c < 259) v = (_Float16)lxyz[c - 256];
                A1[r * KP1 + c] = v;
            }
            if (valid) {
                // 256-half feature row = 32 x 16B segments, 4 per thread, via
                // CDNA5 async global->LDS DMA (tracked by ASYNCcnt)
                const _Float16* src = feat16 + (size_t)idx * DF;
#pragma unroll
                for (int j = 0; j < 4; ++j) {
                    int seg = g * 4 + j;
                    unsigned long long ga = (unsigned long long)(const void*)(src + seg * 8);
                    unsigned int la = (unsigned int)(r * KP1 * 2 + seg * 16);
                    asm volatile("global_load_async_to_lds_b128 %0, %1, off"
                                 :: "v"(la), "v"(ga) : "memory");
                }
            } else {
                // zero the feature region of invalid rows (32 halves per thread)
                for (int c = g * 32; c < g * 32 + 32; ++c)
                    A1[r * KP1 + c] = (_Float16)0.f;
            }
        }
        asm volatile("s_wait_asynccnt 0x0" ::: "memory");
        __syncthreads();
        // layer 1: G1 = relu(A1 @ Wp1 + bp1); wave covers 32 rows x 64 cols (2x2 blocked)
        for (int np = 0; np < 2; ++np) {
            int nt0 = wv * 4 + np * 2;           // ntile pair {nt0, nt0+1}
            int n0 = nt0 * 16 + nl, n1 = n0 + 16;
            float bv0 = bp1[n0], bv1 = bp1[n1];
            v8f acc00, acc01, acc10, acc11;
#pragma unroll
            for (int r = 0; r < 8; ++r) { acc00[r] = bv0; acc01[r] = bv1; acc10[r] = bv0; acc11[r] = bv1; }
            const _Float16* Bb = Wp1t + (size_t)(nt0 * 16) * KP1;
#pragma unroll
            for (int kt = 0; kt < KP1 / 32; ++kt) {
                v16h a0 = load_a16(A1 + kt * 32, KP1);
                v16h a1 = load_a16(A1 + 16 * KP1 + kt * 32, KP1);
                v16h b0 = load_b16(Bb + kt * 32, KP1);
                v16h b1 = load_b16(Bb + 16 * KP1 + kt * 32, KP1);
                acc00 = wmma_f16(a0, b0, acc00);
                acc01 = wmma_f16(a0, b1, acc01);
                acc10 = wmma_f16(a1, b0, acc10);
                acc11 = wmma_f16(a1, b1, acc11);
            }
#pragma unroll
            for (int r = 0; r < 8; ++r) {
                int m0 = r + 8 * hh, m1 = m0 + 16;
                G1[m0 * HID + n0] = (_Float16)fmaxf(acc00[r], 0.f);
                G1[m0 * HID + n1] = (_Float16)fmaxf(acc01[r], 0.f);
                G1[m1 * HID + n0] = (_Float16)fmaxf(acc10[r], 0.f);
                G1[m1 * HID + n1] = (_Float16)fmaxf(acc11[r], 0.f);
            }
        }
        __syncthreads();
        // layer 2 + masked max pool: relu(G1 @ Wp2 + bp2)
        for (int np = 0; np < 2; ++np) {
            int nt0 = wv * 4 + np * 2;
            int n0 = nt0 * 16 + nl, n1 = n0 + 16;
            float bv0 = bp2[n0], bv1 = bp2[n1];
            v8f acc00, acc01, acc10, acc11;
#pragma unroll
            for (int r = 0; r < 8; ++r) { acc00[r] = bv0; acc01[r] = bv1; acc10[r] = bv0; acc11[r] = bv1; }
            const _Float16* Bb = Wp2t + (size_t)(nt0 * 16) * HID;
#pragma unroll
            for (int kt = 0; kt < HID / 32; ++kt) {
                v16h a0 = load_a16(G1 + kt * 32, HID);
                v16h a1 = load_a16(G1 + 16 * HID + kt * 32, HID);
                v16h b0 = load_b16(Bb + kt * 32, HID);
                v16h b1 = load_b16(Bb + 16 * HID + kt * 32, HID);
                acc00 = wmma_f16(a0, b0, acc00);
                acc01 = wmma_f16(a0, b1, acc01);
                acc10 = wmma_f16(a1, b0, acc10);
                acc11 = wmma_f16(a1, b1, acc11);
            }
#pragma unroll
            for (int r = 0; r < 8; ++r) {
                int m0 = r + 8 * hh, m1 = m0 + 16;
                int row0 = ch * MCH + m0, row1 = ch * MCH + m1;
                if (row0 < cnt) {   // relu output >= 0 -> int-ordered atomic max OK
                    atomicMax((int*)&pool[n0], __float_as_int(fmaxf(acc00[r], 0.f)));
                    atomicMax((int*)&pool[n1], __float_as_int(fmaxf(acc01[r], 0.f)));
                }
                if (row1 < cnt) {
                    atomicMax((int*)&pool[n0], __float_as_int(fmaxf(acc10[r], 0.f)));
                    atomicMax((int*)&pool[n1], __float_as_int(fmaxf(acc11[r], 0.f)));
                }
            }
        }
    }
    __syncthreads();
    pooledOut[a * HID + t] = pool[t];
    pooledOut[a * HID + t + 256] = pool[t + 256];
}

// ---------------- heads: score / reg / bin-cls -> proposals ----------------
__global__ void k_heads(const float* __restrict__ pooled, const float* __restrict__ anc_xyz,
                        const float* __restrict__ Wsc, const float* __restrict__ bsc,
                        const float* __restrict__ Wrg, const float* __restrict__ brg,
                        const float* __restrict__ Wcl, const float* __restrict__ bcl,
                        float* __restrict__ prop, float* __restrict__ pscore) {
    int a = blockIdx.x, t = threadIdx.x;
    __shared__ float vals[20];
    if (t < 20) {
        float acc = 0.f;
        const float* p = pooled + a * HID;
        for (int k = 0; k < HID; ++k) {
            float w;
            if (t == 0)      w = Wsc[k];
            else if (t < 8)  w = Wrg[k * 7 + (t - 1)];
            else             w = Wcl[k * NBINS + (t - 8)];
            acc += p[k] * w;
        }
        vals[t] = acc;
    }
    __syncthreads();
    if (t == 0) {
        pscore[a] = 1.f / (1.f + expf(-(vals[0] + bsc[0])));
        float reg[7];
        for (int j = 0; j < 7; ++j) reg[j] = vals[1 + j] + brg[j];
        int best = 0; float bl = vals[8] + bcl[0];
        for (int b = 1; b < NBINS; ++b) {
            float lv = vals[8 + b] + bcl[b];
            if (lv > bl) { bl = lv; best = b; }
        }
        float ry = best * (2.0f * 3.14159265358979323846f / NBINS);
        prop[a * 7 + 0] = anc_xyz[a * 4 + 0] + reg[0];
        prop[a * 7 + 1] = anc_xyz[a * 4 + 1] + reg[1];
        prop[a * 7 + 2] = anc_xyz[a * 4 + 2] + reg[2];
        prop[a * 7 + 3] = reg[3];
        prop[a * 7 + 4] = reg[4];
        prop[a * 7 + 5] = reg[5];
        prop[a * 7 + 6] = reg[6] + ry;
    }
}

// ---------------- stable descending rank sort of 128 proposals ----------------
__global__ void k_sort(const float* __restrict__ prop, const float* __restrict__ pscore,
                       float* __restrict__ out) {
    int a = threadIdx.x;
    float s = pscore[a];
    int rank = 0;
    for (int j = 0; j < NANC; ++j) {
        float sj = pscore[j];
        if (sj > s || (sj == s && j < a)) ++rank;
    }
    for (int c = 0; c < 7; ++c) out[rank * 7 + c] = prop[a * 7 + c];
}

extern "C" void kernel_launch(void* const* d_in, const int* in_sizes, int n_in,
                              void* d_out, int out_size, void* d_ws, size_t ws_size,
                              hipStream_t stream) {
    const float* points = (const float*)d_in[0];
    const float* W1  = (const float*)d_in[1];
    const float* b1  = (const float*)d_in[2];
    const float* W2  = (const float*)d_in[3];
    const float* b2  = (const float*)d_in[4];
    const float* Ws  = (const float*)d_in[5];
    const float* bs  = (const float*)d_in[6];
    const float* Wp1 = (const float*)d_in[7];
    const float* bp1 = (const float*)d_in[8];
    const float* Wp2 = (const float*)d_in[9];
    const float* bp2 = (const float*)d_in[10];
    const float* Wsc = (const float*)d_in[11];
    const float* bsc = (const float*)d_in[12];
    const float* Wrg = (const float*)d_in[13];
    const float* brg = (const float*)d_in[14];
    const float* Wcl = (const float*)d_in[15];
    const float* bcl = (const float*)d_in[16];

    float* prop_out = (float*)d_out;             // (128,7)
    float* feat_out = (float*)d_out + NANC * 7;  // (65536,256) f32

    char* ws = (char*)d_ws;
    size_t off = 0;
    auto carve = [&](size_t bytes) { char* p = ws + off; off = (off + bytes + 255) & ~size_t(255); return p; };
    _Float16* h16    = (_Float16*)carve((size_t)NPTS * DF * 2);
    _Float16* feat16 = (_Float16*)carve((size_t)NPTS * DF * 2);
    _Float16* W2t    = (_Float16*)carve((size_t)DF * DF * 2);
    _Float16* Wp1t   = (_Float16*)carve((size_t)HID * KP1 * 2);
    _Float16* Wp2t   = (_Float16*)carve((size_t)HID * HID * 2);
    float*    score  = (float*)carve((size_t)NPTS * 4);
    int*      ancIdx = (int*)carve((size_t)NANC * 4);
    float*    ancXyz = (float*)carve((size_t)NANC * 4 * 4);
    int*      nidx   = (int*)carve((size_t)NANC * KN * 4);
    int*      ncnt   = (int*)carve((size_t)NANC * 4);
    float*    pooled = (float*)carve((size_t)NANC * HID * 4);
    float*    propT  = (float*)carve((size_t)NANC * 7 * 4);
    float*    pscore = (float*)carve((size_t)NANC * 4);

    // 1) weight prep (f16, column-major; Wp1 K-permuted to match A layout)
    k_prep_wt<<<DF, 128, 0, stream>>>(W2, W2t, DF, DF, DF);
    k_prep_wp1<<<HID, 128, 0, stream>>>(Wp1, Wp1t);
    k_prep_wt<<<HID, 128, 0, stream>>>(Wp2, Wp2t, HID, HID, HID);

    // 2) backbone
    k_h<<<NPTS, DF, 0, stream>>>(points, W1, b1, h16);
    k_feat<<<(NPTS / 32) * (DF / 32) / 8, 256, 0, stream>>>(h16, W2t, b2, feat_out, feat16);
    k_score<<<NPTS / 256, 256, 0, stream>>>(feat_out, Ws, bs, score);

    // 3) NMS anchor selection (sequential by construction)
    k_nms<<<1, 1024, 0, stream>>>(score, points, ancIdx, ancXyz);

    // 4) neighbor selection
    k_select<<<NANC, 256, 0, stream>>>(points, ancXyz, nidx, ncnt);

    // 5) per-anchor MLP + max pool (WMMA + async LDS gather)
    size_t lds = (size_t)MCH * KP1 * 2 + (size_t)MCH * HID * 2 + (size_t)HID * 4;
    k_mlp<<<NANC, 256, lds, stream>>>(points, feat16, ancXyz, nidx, ncnt,
                                      Wp1t, bp1, Wp2t, bp2, pooled);

    // 6) heads + proposal assembly + stable sort
    k_heads<<<NANC, 32, 0, stream>>>(pooled, ancXyz, Wsc, bsc, Wrg, brg, Wcl, bcl,
                                     propT, pscore);
    k_sort<<<1, NANC, 0, stream>>>(propT, pscore, prop_out);
}
